// PPFConv_43087111914330
// MI455X (gfx1250) — compile-verified, matrix operand order checked by reading, not run on hardware
//
#include <hip/hip_runtime.h>
#include <hip/hip_bf16.h>

typedef __attribute__((ext_vector_type(16))) _Float16     v16h;
typedef __attribute__((ext_vector_type(8)))  float        v8f;
typedef __attribute__((ext_vector_type(8)))  int          v8i;
typedef __attribute__((ext_vector_type(4)))  unsigned int v4u;
typedef __attribute__((ext_vector_type(8)))  int          v8i32;
typedef __attribute__((ext_vector_type(4)))  int          v4i32;

#define F_IN 64
#define HDIM 64
#define ODIM 128
#define K1PAD 100   // 68 real K, compute over 96 (3 x K32); stride 100 f16 = 50 dw (rows hit distinct banks)
#define K2PAD 72    // 64 real K; stride 72 f16 = 36 dw (rows hit distinct banks)
#define TILE_E 256
#define NB_SUM 1024

static __device__ __forceinline__ unsigned short h2us(_Float16 h) {
    return __builtin_bit_cast(unsigned short, h);
}

// Assemble one 16x32 f16 WMMA operand fragment from LDS (row-major, rowDw = row base in dwords).
// Per CDNA5 ISA 7.12.2: low-half lanes hold K pairs {0..7,16..23}(+kb); high-half lanes kb+8.
static __device__ __forceinline__ v16h ldfrag(const unsigned int* s, int rowDw, int kDw) {
    const unsigned int* p = s + rowDw + kDw;
    v8i t;
    t[0] = p[0];  t[1] = p[1];  t[2] = p[2];  t[3] = p[3];
    t[4] = p[8];  t[5] = p[9];  t[6] = p[10]; t[7] = p[11];
    return __builtin_bit_cast(v16h, t);
}

static __device__ __forceinline__ float angle3(float ax, float ay, float az,
                                               float bx, float by, float bz) {
    float cx = ay * bz - az * by;
    float cy = az * bx - ax * bz;
    float cz = ax * by - ay * bx;
    return atan2f(sqrtf(cx * cx + cy * cy + cz * cz), ax * bx + ay * by + az * bz);
}

// ---- Tensor Data Mover: 1-D contiguous fp16 copy global -> LDS (D# per ISA 8.3/8.4) ----
// Issued once per block by wave 0; tracked with TENSORcnt.
static __device__ __forceinline__ void tdm_load_1d_f16(unsigned lds_off, const void* gptr,
                                                       unsigned nelem) {
    unsigned long long ga = (unsigned long long)gptr;
    v4u g0;
    g0[0] = 1u;                                            // count=1, is_restore=0, gather off
    g0[1] = lds_off;                                       // lds_addr (bytes)
    g0[2] = (unsigned)ga;                                  // global_addr[31:0]
    g0[3] = (unsigned)((ga >> 32) & 0x1FFFFFFu) | (2u << 30); // global_addr[56:32] | type=2
    v8i32 g1;
    g1[0] = (int)(1u << 16);                               // workgroup_mask=0, data_size=2B
    g1[1] = (int)((nelem & 0xFFFFu) << 16);                // tensor_dim0[15:0] @ bits63:48
    g1[2] = (int)((nelem >> 16) | (1u << 16));             // tensor_dim0[31:16] | tensor_dim1=1
    g1[3] = (int)((nelem & 0xFFFFu) << 16);                // tile_dim0 @ bits127:112
    g1[4] = 1;                                             // tile_dim1=1, tile_dim2=0
    g1[5] = (int)nelem;                                    // tensor_dim0_stride[31:0]
    g1[6] = 0;                                             // stride hi, dim1_stride
    g1[7] = 0;
    v4i32 z4 = {0, 0, 0, 0};
#if defined(__clang_major__) && __clang_major__ >= 23
    v8i32 z8 = {0, 0, 0, 0, 0, 0, 0, 0};
    __builtin_amdgcn_tensor_load_to_lds(g0, g1, z4, z4, z8, 0);
#else
    __builtin_amdgcn_tensor_load_to_lds(g0, g1, z4, z4, 0);
#endif
}

// ---------------- init kernels ----------------

__global__ __launch_bounds__(256) void k_zero_u32(unsigned int* p, long n) {
    for (long i = (long)blockIdx.x * 256 + threadIdx.x; i < n; i += (long)gridDim.x * 256)
        p[i] = 0u;
}

__global__ __launch_bounds__(256) void k_cvt_x(const float* x, _Float16* xh, long n) {
    for (long i = (long)blockIdx.x * 256 + threadIdx.x; i < n; i += (long)gridDim.x * 256)
        xh[i] = (_Float16)x[i];
}

__global__ __launch_bounds__(256) void k_build_w1t(const float* W1, _Float16* w1t) {
    for (int i = blockIdx.x * 256 + threadIdx.x; i < HDIM * K1PAD; i += gridDim.x * 256) {
        int n = i / K1PAD, k = i % K1PAD;
        w1t[i] = (k < F_IN + 4) ? (_Float16)W1[k * HDIM + n] : (_Float16)0.0f;
    }
}

__global__ __launch_bounds__(256) void k_build_w2t(const float* W2, _Float16* w2t) {
    for (int i = blockIdx.x * 256 + threadIdx.x; i < ODIM * K2PAD; i += gridDim.x * 256) {
        int n = i / K2PAD, k = i % K2PAD;
        w2t[i] = (k < HDIM) ? (_Float16)W2[k * ODIM + n] : (_Float16)0.0f;
    }
}

// ---------------- mean squared distance (deterministic two-pass) ----------------

__global__ __launch_bounds__(256) void k_sum_partial(const int* ei, const float* pos,
                                                     float* partials, int E) {
    __shared__ float red[256];
    float loc = 0.0f;
    for (long e = (long)blockIdx.x * 256 + threadIdx.x; e < E; e += (long)NB_SUM * 256) {
        int r = ei[e], c = ei[E + e];
        float dx = pos[3 * c + 0] - pos[3 * r + 0];
        float dy = pos[3 * c + 1] - pos[3 * r + 1];
        float dz = pos[3 * c + 2] - pos[3 * r + 2];
        loc += dx * dx + dy * dy + dz * dz;
    }
    red[threadIdx.x] = loc;
    __syncthreads();
    for (int s = 128; s > 0; s >>= 1) {
        if ((int)threadIdx.x < s) red[threadIdx.x] += red[threadIdx.x + s];
        __syncthreads();
    }
    if (threadIdx.x == 0) partials[blockIdx.x] = red[0];
}

__global__ __launch_bounds__(256) void k_sum_final(const float* partials, float* sumv) {
    __shared__ float red[256];
    float loc = 0.0f;
    for (int i = threadIdx.x; i < NB_SUM; i += 256) loc += partials[i];
    red[threadIdx.x] = loc;
    __syncthreads();
    for (int s = 128; s > 0; s >>= 1) {
        if ((int)threadIdx.x < s) red[threadIdx.x] += red[threadIdx.x + s];
        __syncthreads();
    }
    if (threadIdx.x == 0) sumv[0] = red[0];
}

// ---------------- edge kernel: ppf + gather + WMMA GEMM1 + atomic scatter-max ----------------

__global__ __launch_bounds__(256) void k_edges(const int* ei, const float* pos, const float* nrm,
                                               const _Float16* xh, const _Float16* w1t,
                                               const float* b1, const float* sumv,
                                               unsigned int* acc, int N, int E) {
    __shared__ unsigned int AshI[TILE_E * (K1PAD / 2)];   // 256 x 100 f16 = 51200 B
    __shared__ unsigned int BshI[HDIM * (K1PAD / 2)];     // 64  x 100 f16 = 12800 B
    __shared__ float b1s[HDIM];
    __shared__ int rows_s[TILE_E];

    const int t = threadIdx.x;
    const int lane = t & 31, wave = t >> 5;
    const int Etot = E + N;
    const float invMean = (float)Etot / sumv[0];

    // stage W1^T via the Tensor Data Mover (one DMA op, issued by wave 0)
    if (wave == 0) {
        tdm_load_1d_f16((unsigned)(unsigned long long)BshI, w1t, HDIM * K1PAD);
    }
    if (t < HDIM) b1s[t] = b1[t];

    // stage one edge per thread into the A tile
    {
        const int e = blockIdx.x * TILE_E + t;
        unsigned int* arow = AshI + t * (K1PAD / 2);
        if (e < Etot) {
            int r, c;
            if (e < E) { r = ei[e]; c = ei[E + e]; }
            else       { r = e - E; c = e - E; }
            rows_s[t] = r;
            float prx = pos[3 * r], pry = pos[3 * r + 1], prz = pos[3 * r + 2];
            float pcx = pos[3 * c], pcy = pos[3 * c + 1], pcz = pos[3 * c + 2];
            float nrx = nrm[3 * r], nry = nrm[3 * r + 1], nrz = nrm[3 * r + 2];
            float ncx = nrm[3 * c], ncy = nrm[3 * c + 1], ncz = nrm[3 * c + 2];
            float dx = pcx - prx, dy = pcy - pry, dz = pcz - prz;
            float dist = (dx * dx + dy * dy + dz * dz) * invMean;
            float a1 = angle3(nrx, nry, nrz, dx, dy, dz);
            float a2 = angle3(ncx, ncy, ncz, dx, dy, dz);
            float a3 = angle3(nrx, nry, nrz, ncx, ncy, ncz);
            // x[col] gather: 64 fp16 = 128 B via 8x B128 loads (source 128B-aligned)
            const uint4* xs = (const uint4*)(xh + (long)c * F_IN);
            uint2* ad = (uint2*)arow;   // LDS row is 8B-aligned (t*200)
            #pragma unroll
            for (int j = 0; j < 8; ++j) {
                uint4 v = xs[j];
                ad[2 * j + 0] = make_uint2(v.x, v.y);
                ad[2 * j + 1] = make_uint2(v.z, v.w);
            }
            arow[32] = (unsigned int)h2us((_Float16)dist) | ((unsigned int)h2us((_Float16)a1) << 16);
            arow[33] = (unsigned int)h2us((_Float16)a2)   | ((unsigned int)h2us((_Float16)a3) << 16);
            #pragma unroll
            for (int j = 34; j < K1PAD / 2; ++j) arow[j] = 0u;
        } else {
            rows_s[t] = -1;
            #pragma unroll
            for (int j = 0; j < K1PAD / 2; ++j) arow[j] = 0u;
        }
    }
    if (wave == 0) __builtin_amdgcn_s_wait_tensorcnt((short)0);
    __syncthreads();

    // WMMA: each wave owns 2 M-tiles; A fragments + row ids cached in registers,
    // B fragments streamed per N-tile (3 K-steps each).
    const int l15 = lane & 15;
    const int khalf = (lane < 16) ? 0 : 4;   // dword offset of lane's K half
    const int mhi = (lane < 16) ? 0 : 8;
    const bool full = (blockIdx.x + 1) * TILE_E <= Etot;   // uniform fast path
    const int mtb = wave * 2;

    v16h af[2][3];
    #pragma unroll
    for (int m2 = 0; m2 < 2; ++m2)
        #pragma unroll
        for (int ks = 0; ks < 3; ++ks)
            af[m2][ks] = ldfrag(AshI, ((mtb + m2) * 16 + l15) * (K1PAD / 2), ks * 16 + khalf);

    int rs[2][8];
    #pragma unroll
    for (int m2 = 0; m2 < 2; ++m2) {
        const int4* rp = (const int4*)&rows_s[(mtb + m2) * 16 + mhi];  // 32B aligned
        int4 rA = rp[0], rB = rp[1];
        rs[m2][0] = rA.x; rs[m2][1] = rA.y; rs[m2][2] = rA.z; rs[m2][3] = rA.w;
        rs[m2][4] = rB.x; rs[m2][5] = rB.y; rs[m2][6] = rB.z; rs[m2][7] = rB.w;
    }

    for (int nt = 0; nt < 4; ++nt) {
        const int nrow = (nt * 16 + l15) * (K1PAD / 2);
        v16h bf0 = ldfrag(BshI, nrow, 0 + khalf);
        v16h bf1 = ldfrag(BshI, nrow, 16 + khalf);
        v16h bf2 = ldfrag(BshI, nrow, 32 + khalf);
        const int col = nt * 16 + l15;
        const float bb = b1s[col];
        #pragma unroll
        for (int m2 = 0; m2 < 2; ++m2) {
            v8f c = {};
            c = __builtin_amdgcn_wmma_f32_16x16x32_f16(false, af[m2][0], false, bf0, (short)0, c, false, false);
            c = __builtin_amdgcn_wmma_f32_16x16x32_f16(false, af[m2][1], false, bf1, (short)0, c, false, false);
            c = __builtin_amdgcn_wmma_f32_16x16x32_f16(false, af[m2][2], false, bf2, (short)0, c, false, false);
            if (full) {
                #pragma unroll
                for (int v = 0; v < 8; ++v) {
                    float val = fmaxf(c[v] + bb, 0.0f);   // >=0: int-bit atomicMax == float max
                    atomicMax((int*)acc + (rs[m2][v] * HDIM + col), __float_as_int(val));
                }
            } else {
                #pragma unroll
                for (int v = 0; v < 8; ++v) {
                    const int r = rs[m2][v];
                    if (r >= 0) {
                        float val = fmaxf(c[v] + bb, 0.0f);
                        atomicMax((int*)acc + (r * HDIM + col), __float_as_int(val));
                    }
                }
            }
        }
    }
}

// ---------------- node kernel: WMMA GEMM2 + ReLU ----------------

__global__ __launch_bounds__(256) void k_nodes(const unsigned int* acc, const _Float16* w2t,
                                               const float* b2, float* out, int N) {
    __shared__ unsigned int A2I[128 * (K2PAD / 2)];   // 128 x 72 f16 = 18432 B
    __shared__ unsigned int W2I[ODIM * (K2PAD / 2)];  // 128 x 72 f16 = 18432 B
    __shared__ float b2s[ODIM];

    const int t = threadIdx.x;
    const int lane = t & 31, wave = t >> 5;
    const int r0 = blockIdx.x * 128;

    if (wave == 0) {
        tdm_load_1d_f16((unsigned)(unsigned long long)W2I, w2t, ODIM * K2PAD);
    }
    for (int i = t; i < 128 * F_IN; i += 256) {
        int row = i >> 6, k = i & 63;
        long node = (long)r0 + row;
        float v = (node < N) ? __int_as_float((int)acc[node * HDIM + k]) : 0.0f;
        ((_Float16*)A2I)[row * K2PAD + k] = (_Float16)v;
    }
    for (int i = t; i < 128 * (K2PAD - F_IN); i += 256) {
        int row = i >> 3, j = i & 7;
        ((_Float16*)A2I)[row * K2PAD + F_IN + j] = (_Float16)0.0f;
    }
    if (t < ODIM) b2s[t] = b2[t];
    if (wave == 0) __builtin_amdgcn_s_wait_tensorcnt((short)0);
    __syncthreads();

    const int l15 = lane & 15;
    const int khalf = (lane < 16) ? 0 : 4;
    const int mhi = (lane < 16) ? 0 : 8;
    const bool full = r0 + 128 <= N;

    v16h af0 = ldfrag(A2I, (wave * 16 + l15) * (K2PAD / 2), 0 + khalf);
    v16h af1 = ldfrag(A2I, (wave * 16 + l15) * (K2PAD / 2), 16 + khalf);

    for (int nt = 0; nt < 8; ++nt) {
        const int nrow = (nt * 16 + l15) * (K2PAD / 2);
        v16h bf0 = ldfrag(W2I, nrow, 0 + khalf);
        v16h bf1 = ldfrag(W2I, nrow, 16 + khalf);
        v8f c = {};
        c = __builtin_amdgcn_wmma_f32_16x16x32_f16(false, af0, false, bf0, (short)0, c, false, false);
        c = __builtin_amdgcn_wmma_f32_16x16x32_f16(false, af1, false, bf1, (short)0, c, false, false);
        const int col = nt * 16 + l15;
        const float bb = b2s[col];
        if (full) {
            #pragma unroll
            for (int v = 0; v < 8; ++v) {
                int node = r0 + wave * 16 + v + mhi;
                out[node * ODIM + col] = fmaxf(c[v] + bb, 0.0f);
            }
        } else {
            #pragma unroll
            for (int v = 0; v < 8; ++v) {
                int node = r0 + wave * 16 + v + mhi;
                if (node < N) out[node * ODIM + col] = fmaxf(c[v] + bb, 0.0f);
            }
        }
    }
}

// ---------------- host launcher ----------------

extern "C" void kernel_launch(void* const* d_in, const int* in_sizes, int n_in,
                              void* d_out, int out_size, void* d_ws, size_t ws_size,
                              hipStream_t stream) {
    const float* x   = (const float*)d_in[0];
    const float* pos = (const float*)d_in[1];
    const float* nrm = (const float*)d_in[2];
    const int*   ei  = (const int*)d_in[3];
    // d_in[4] = batch (all zeros, unused: single graph)
    const float* W1  = (const float*)d_in[5];
    const float* b1  = (const float*)d_in[6];
    const float* W2  = (const float*)d_in[7];
    const float* b2  = (const float*)d_in[8];
    float* out = (float*)d_out;

    const int N = in_sizes[0] / F_IN;
    const int E = in_sizes[3] / 2;
    const int Etot = E + N;

    char* ws = (char*)d_ws;
    float*        sumv     = (float*)ws;                                   // 1 float
    float*        partials = (float*)(ws + 256);                           // NB_SUM floats
    unsigned int* acc      = (unsigned int*)(ws + 4608);                   // N*64 u32
    _Float16*     xh       = (_Float16*)(ws + 4608 + (size_t)N * 256);     // N*64 f16
    _Float16*     w1t      = (_Float16*)(ws + 4608 + (size_t)N * 256 + (size_t)N * 128);
    _Float16*     w2t      = w1t + HDIM * K1PAD;

    // init
    k_zero_u32<<<2048, 256, 0, stream>>>(acc, (long)N * HDIM);
    k_cvt_x<<<2048, 256, 0, stream>>>(x, xh, (long)N * F_IN);
    k_build_w1t<<<(HDIM * K1PAD + 255) / 256, 256, 0, stream>>>(W1, w1t);
    k_build_w2t<<<(ODIM * K2PAD + 255) / 256, 256, 0, stream>>>(W2, w2t);

    // mean squared distance (self-loops contribute 0 to the sum; Etot in denominator)
    k_sum_partial<<<NB_SUM, 256, 0, stream>>>(ei, pos, partials, E);
    k_sum_final<<<1, 256, 0, stream>>>(partials, sumv);

    // GEMM1 + segment-max
    k_edges<<<(Etot + TILE_E - 1) / TILE_E, 256, 0, stream>>>(
        ei, pos, nrm, xh, w1t, b1, sumv, acc, N, E);

    // GEMM2
    k_nodes<<<(N + 127) / 128, 256, 0, stream>>>(acc, w2t, b2, out, N);
}
